// RecurrentReadout_27728308863499
// MI455X (gfx1250) — matrix operand order repaired and checked by reading, not compiled, not run on hardware
//
#include <hip/hip_runtime.h>
#include <hip/hip_bf16.h>
#include <stdint.h>

// Problem constants (from reference)
#define BB   16
#define TT   256
#define CC   1024
#define HW   64          // 8*8 spatial
#define OUTN 1024
#define MM   (BB * TT)   // 4096

typedef __attribute__((ext_vector_type(16))) __bf16 v16bf;
typedef __attribute__((ext_vector_type(8)))  float  v8f;

union Frag {
    unsigned int u[8];
    v16bf v;
};

// ---------------------------------------------------------------------------
// Stage 1: spatial mean of 64 contiguous floats -> bf16.
// 16 lanes cooperate per output: lane j loads one float4; a full wave32 reads
// 512 contiguous bytes per iteration (perfect coalescing for the 1 GiB stream).
// ---------------------------------------------------------------------------
__global__ __launch_bounds__(256) void spatial_mean_kernel(
    const float* __restrict__ x, __hip_bfloat16* __restrict__ xm, int n_out)
{
    const int tid    = blockIdx.x * blockDim.x + threadIdx.x;
    const int lane16 = threadIdx.x & 15;
    const int groups = (gridDim.x * blockDim.x) >> 4;

    for (int out = tid >> 4; out < n_out; out += groups) {
        const float4* p = (const float4*)(x + (size_t)out * HW);
        float4 v = p[lane16];
        float  s = (v.x + v.y) + (v.z + v.w);
        s += __shfl_xor(s, 1, 32);
        s += __shfl_xor(s, 2, 32);
        s += __shfl_xor(s, 4, 32);
        s += __shfl_xor(s, 8, 32);
        if (lane16 == 0)
            xm[out] = __float2bfloat16(s * (1.0f / (float)HW));
    }
}

// ---------------------------------------------------------------------------
// Tiny fp32 -> bf16 conversion for W (2 MB result, lives in L2 afterwards)
// ---------------------------------------------------------------------------
__global__ __launch_bounds__(256) void convert_w_kernel(
    const float* __restrict__ W, __hip_bfloat16* __restrict__ Wb, int n4)
{
    int i = blockIdx.x * blockDim.x + threadIdx.x;
    if (i < n4) {
        float4 v = ((const float4*)W)[i];
        __hip_bfloat16* o = Wb + (size_t)i * 4;
        o[0] = __float2bfloat16(v.x);
        o[1] = __float2bfloat16(v.y);
        o[2] = __float2bfloat16(v.z);
        o[3] = __float2bfloat16(v.w);
    }
}

// ---------------------------------------------------------------------------
// Stage 2: U = xm(bf16) @ W(bf16)^T + b, fp32 accumulate via
// v_wmma_f32_16x16x32_bf16.
//   Block = 256 threads = 8 waves, block tile 128(M) x 64(N), BK = 32.
//   Wave (wm,wn) = (wave>>1, wave&1) computes a 32x32 patch:
//   2 A-frags x 2 B-frags -> 4 WMMAs per K-step.
//   Double-buffered LDS filled by GLOBAL_LOAD_ASYNC_TO_LDS_B128 (ASYNCcnt):
//   no VGPR round-trip, one s_wait_asynccnt + one barrier per K-step.
//   LDS rows padded to 34 bf16 (odd dword stride) -> near-conflict-free gathers.
// ---------------------------------------------------------------------------
#define LDS_STRIDE 34

__global__ __launch_bounds__(256) void gemm_bf16_wmma_kernel(
    const __hip_bfloat16* __restrict__ A,   // xm  [MM, CC] row-major
    const __hip_bfloat16* __restrict__ Bw,  // Wb  [OUTN, CC] row-major (N,K)
    const float* __restrict__ bias,         // [OUTN]
    float* __restrict__ U)                  // [MM, OUTN]
{
    __shared__ __align__(16) __hip_bfloat16 lA[2][128 * LDS_STRIDE]; // 2 x 8704 B
    __shared__ __align__(16) __hip_bfloat16 lB[2][ 64 * LDS_STRIDE]; // 2 x 4352 B

    const int tid  = threadIdx.x;
    const int wave = tid >> 5;
    const int lane = tid & 31;
    const int l    = lane & 15;
    const int hi   = lane >> 4;

    const int m0 = (blockIdx.x >> 4) * 128;  // 32 M-tiles of 128
    const int n0 = (blockIdx.x & 15) * 64;   // 16 N-tiles of 64
    const int wm = (wave >> 1) * 32;         // wave M offset in block tile
    const int wn = (wave & 1) * 32;          // wave N offset in block tile

    const int r  = tid >> 2;                 // 0..63
    const int c8 = (tid & 3) * 8;            // 0,8,16,24 (bf16 elements)

    const __hip_bfloat16* gA0 = A  + (size_t)(m0 + r)      * CC + c8;
    const __hip_bfloat16* gA1 = A  + (size_t)(m0 + r + 64) * CC + c8;
    const __hip_bfloat16* gB  = Bw + (size_t)(n0 + r)      * CC + c8;

    // Issue the 3 async 16-byte global->LDS copies for K-chunk `kk` into buffer b.
    auto stage = [&](int kk, int b) {
        unsigned da0 = (unsigned)(uintptr_t)&lA[b][ r       * LDS_STRIDE + c8];
        unsigned da1 = (unsigned)(uintptr_t)&lA[b][(r + 64) * LDS_STRIDE + c8];
        unsigned db  = (unsigned)(uintptr_t)&lB[b][ r       * LDS_STRIDE + c8];
        const void* pa0 = (const void*)(gA0 + kk);
        const void* pa1 = (const void*)(gA1 + kk);
        const void* pb  = (const void*)(gB  + kk);
        asm volatile("global_load_async_to_lds_b128 %0, %1, off"
                     :: "v"(da0), "v"(pa0) : "memory");
        asm volatile("global_load_async_to_lds_b128 %0, %1, off"
                     :: "v"(da1), "v"(pa1) : "memory");
        asm volatile("global_load_async_to_lds_b128 %0, %1, off"
                     :: "v"(db),  "v"(pb)  : "memory");
    };

    v8f acc00 = {};
    v8f acc01 = {};
    v8f acc10 = {};
    v8f acc11 = {};

    // Prologue: fill buffer 0 with chunk 0.
    stage(0, 0);
    asm volatile("s_wait_asynccnt 0x0" ::: "memory");
    __syncthreads();

    for (int kk = 0; kk < CC; kk += 32) {
        const int buf = (kk >> 5) & 1;
        if (kk + 32 < CC)
            stage(kk + 32, buf ^ 1);   // overlap next chunk with this chunk's math

        // Build fragments per the CDNA5 VGPR layout tables.
        Frag fa0, fa1, fb0, fb1;
#pragma unroll
        for (int j = 0; j < 8; ++j) {
            // A 16x32 bf16: lane half 'hi' holds K {0-7,16-23} / {8-15,24-31}
            const int k0a = 2 * (j & 3) + 16 * (j >> 2) + 8 * hi;
            fa0.u[j] = *(const unsigned int*)&lA[buf][(wm +  0 + l) * LDS_STRIDE + k0a];
            fa1.u[j] = *(const unsigned int*)&lA[buf][(wm + 16 + l) * LDS_STRIDE + k0a];
            // B 32x16 bf16: lane half 'hi' holds K {0-7,16-23} / {8-15,24-31}, lane = N
            const int k0b = 2 * (j & 3) + 16 * (j >> 2) + 8 * hi;
            fb0.u[j] = *(const unsigned int*)&lB[buf][(wn +  0 + l) * LDS_STRIDE + k0b];
            fb1.u[j] = *(const unsigned int*)&lB[buf][(wn + 16 + l) * LDS_STRIDE + k0b];
        }

        acc00 = __builtin_amdgcn_wmma_f32_16x16x32_bf16(
                    false, fa0.v, false, fb0.v, (short)0, acc00, false, false);
        acc01 = __builtin_amdgcn_wmma_f32_16x16x32_bf16(
                    false, fa0.v, false, fb1.v, (short)0, acc01, false, false);
        acc10 = __builtin_amdgcn_wmma_f32_16x16x32_bf16(
                    false, fa1.v, false, fb0.v, (short)0, acc10, false, false);
        acc11 = __builtin_amdgcn_wmma_f32_16x16x32_bf16(
                    false, fa1.v, false, fb1.v, (short)0, acc11, false, false);

        // Gathers above feed the WMMAs (dscnt waits precede them), so LDS reads
        // of `buf` are done before this barrier; next iteration refills `buf`.
        asm volatile("s_wait_asynccnt 0x0" ::: "memory");
        __syncthreads();
    }

    // C/D layout: VGPR rr -> M = rr + 8*hi, lane -> N = l
    const int nA = n0 + wn + l;
    const int nB = nA + 16;
    const float bnA = bias[nA];
    const float bnB = bias[nB];
#pragma unroll
    for (int rr = 0; rr < 8; ++rr) {
        const int mA = m0 + wm + rr + 8 * hi;
        const int mB = mA + 16;
        U[(size_t)mA * OUTN + nA] = acc00[rr] + bnA;
        U[(size_t)mA * OUTN + nB] = acc01[rr] + bnB;
        U[(size_t)mB * OUTN + nA] = acc10[rr] + bnA;
        U[(size_t)mB * OUTN + nB] = acc11[rr] + bnB;
    }
}

// ---------------------------------------------------------------------------
// Stage 3: r_t = alpha * r_{t-1} + u_t.  One thread per 4 channels (float4
// loads/stores, 4 independent FMA chains).  Writes r_seq then r_final.
// ---------------------------------------------------------------------------
__global__ __launch_bounds__(256) void recurrence_kernel(
    const float* __restrict__ U,      // [BB, TT, OUTN]
    const float* __restrict__ r0,     // [BB, OUTN]
    const float* __restrict__ alpha,  // [OUTN]
    float* __restrict__ out)          // r_seq [BB,TT,OUTN] ++ r_final [BB,OUTN]
{
    const int tid = blockIdx.x * blockDim.x + threadIdx.x;   // 4096 threads
    const int b   = tid >> 8;            // OUTN/4 = 256 groups per batch
    const int o4  = (tid & 255) * 4;
    if (b >= BB) return;

    float4 rv = *(const float4*)(r0 + (size_t)b * OUTN + o4);
    float4 al = *(const float4*)(alpha + o4);

    const float4* u  = (const float4*)(U   + (size_t)b * TT * OUTN + o4);
    float4*       rs = (float4*)      (out + (size_t)b * TT * OUTN + o4);
    const int stride4 = OUTN >> 2;

#pragma unroll 4
    for (int t = 0; t < TT; ++t) {
        float4 uv = u[(size_t)t * stride4];
        rv.x = fmaf(al.x, rv.x, uv.x);
        rv.y = fmaf(al.y, rv.y, uv.y);
        rv.z = fmaf(al.z, rv.z, uv.z);
        rv.w = fmaf(al.w, rv.w, uv.w);
        rs[(size_t)t * stride4] = rv;
    }

    *(float4*)(out + (size_t)BB * TT * OUTN + (size_t)b * OUTN + o4) = rv;
}

// ---------------------------------------------------------------------------
extern "C" void kernel_launch(void* const* d_in, const int* in_sizes, int n_in,
                              void* d_out, int out_size, void* d_ws, size_t ws_size,
                              hipStream_t stream)
{
    const float* x     = (const float*)d_in[0];  // [16,256,1024,8,8]
    const float* r0    = (const float*)d_in[1];  // [16,1024]
    const float* W     = (const float*)d_in[2];  // [1024,1024]
    const float* bias  = (const float*)d_in[3];  // [1024]
    const float* alpha = (const float*)d_in[4];  // [1024]
    float* out = (float*)d_out;

    char* ws = (char*)d_ws;
    __hip_bfloat16* xm = (__hip_bfloat16*)ws;                             // 8 MB
    __hip_bfloat16* Wb = (__hip_bfloat16*)(ws + (size_t)8 * 1024 * 1024); // 2 MB
    float*          U  = (float*)(ws + (size_t)10 * 1024 * 1024);         // 16 MB

    // Stage 1: 1 GiB streaming mean (bandwidth floor ~46 us at 23.3 TB/s)
    spatial_mean_kernel<<<1024, 256, 0, stream>>>(x, xm, MM * CC);

    // W -> bf16 (1M elements)
    convert_w_kernel<<<1024, 256, 0, stream>>>(W, Wb, (OUTN * CC) / 4);

    // Stage 2: WMMA bf16 GEMM, 32 M-tiles x 16 N-tiles of 128x64
    gemm_bf16_wmma_kernel<<<(MM / 128) * (OUTN / 64), 256, 0, stream>>>(xm, Wb, bias, U);

    // Stage 3: diagonal recurrence, 16*256 float4 channels
    recurrence_kernel<<<(BB * (OUTN / 4) + 255) / 256, 256, 0, stream>>>(U, r0, alpha, out);
}
